// GaussianRasterizer_57329223467372
// MI455X (gfx1250) — matrix operand order repaired and checked by reading, not compiled, not run on hardware
//
#include <hip/hip_runtime.h>
#include <hip/hip_bf16.h>
#include <stdint.h>

#define IMG_W 320
#define IMG_H 240
#define NG    512
#define NPIX  (IMG_W * IMG_H)

// ---- workspace layout (float offsets) ----
#define WS_KEY 0
#define WS_PX  512
#define WS_PY  1024
#define WS_CA  1536
#define WS_CB  2048
#define WS_CC  2560
#define WS_Z   3072
#define WS_RAD 3584
#define WS_VIS 4096
#define WS_SG  4608            // sorted structs: 512 * 3 float4 = 6144 floats
#define WS_BF  10752           // B fragments: 16 chunks * 32 lanes * 8 dwords = 4096

typedef _Float16 v16h __attribute__((ext_vector_type(16)));
typedef float    v8f  __attribute__((ext_vector_type(8)));

union V16U { v16h v; uint32_t u[8]; };

__device__ __forceinline__ uint32_t f2h_bits(float x) {
  _Float16 h = (_Float16)x;
  uint16_t u;
  __builtin_memcpy(&u, &h, 2);
  return (uint32_t)u;
}

// ---------------- Kernel 1: per-gaussian projection / cov2d / visibility ----
__global__ void __launch_bounds__(NG) gs_preproc(
    const float* __restrict__ means, const float* __restrict__ scales,
    const float* __restrict__ rots, const float* __restrict__ Km,
    const float* __restrict__ Tm, float* __restrict__ ws,
    float* __restrict__ radii_out) {
  int i = threadIdx.x;
  if (i >= NG) return;

  const float fx = Km[0], cx = Km[2], fy = Km[4], cy = Km[5];
  float Rc[3][3] = {{Tm[0], Tm[1], Tm[2]},
                    {Tm[4], Tm[5], Tm[6]},
                    {Tm[8], Tm[9], Tm[10]}};
  float tv[3] = {Tm[3], Tm[7], Tm[11]};

  float mx = means[3 * i], my = means[3 * i + 1], mz = means[3 * i + 2];
  float xc = Rc[0][0] * mx + Rc[0][1] * my + Rc[0][2] * mz + tv[0];
  float yc = Rc[1][0] * mx + Rc[1][1] * my + Rc[1][2] * mz + tv[1];
  float zc = Rc[2][0] * mx + Rc[2][1] * my + Rc[2][2] * mz + tv[2];
  float zs = (zc == 0.0f) ? 1e-8f : zc;
  float px = fx * xc / zs + cx;
  float py = fy * yc / zs + cy;

  // quaternion -> rotation
  float qw = rots[4 * i], qx = rots[4 * i + 1], qy = rots[4 * i + 2], qz = rots[4 * i + 3];
  float qn = sqrtf(qw * qw + qx * qx + qy * qy + qz * qz);
  qw /= qn; qx /= qn; qy /= qn; qz /= qn;
  float R[3][3] = {
      {1.0f - 2.0f * (qy * qy + qz * qz), 2.0f * (qx * qy - qw * qz), 2.0f * (qx * qz + qw * qy)},
      {2.0f * (qx * qy + qw * qz), 1.0f - 2.0f * (qx * qx + qz * qz), 2.0f * (qy * qz - qw * qx)},
      {2.0f * (qx * qz - qw * qy), 2.0f * (qy * qz + qw * qx), 1.0f - 2.0f * (qx * qx + qy * qy)}};

  float s0 = scales[3 * i], s1 = scales[3 * i + 1], s2 = scales[3 * i + 2];
  float sq[3] = {s0 * s0, s1 * s1, s2 * s2};

  float cov[3][3];
#pragma unroll
  for (int r = 0; r < 3; ++r)
#pragma unroll
    for (int c = 0; c < 3; ++c)
      cov[r][c] = R[r][0] * sq[0] * R[c][0] + R[r][1] * sq[1] * R[c][1] + R[r][2] * sq[2] * R[c][2];

  // V = Rc * cov * Rc^T
  float Tmp[3][3];
#pragma unroll
  for (int r = 0; r < 3; ++r)
#pragma unroll
    for (int c = 0; c < 3; ++c)
      Tmp[r][c] = Rc[r][0] * cov[0][c] + Rc[r][1] * cov[1][c] + Rc[r][2] * cov[2][c];
  float V[3][3];
#pragma unroll
  for (int r = 0; r < 3; ++r)
#pragma unroll
    for (int c = 0; c < 3; ++c)
      V[r][c] = Tmp[r][0] * Rc[c][0] + Tmp[r][1] * Rc[c][1] + Tmp[r][2] * Rc[c][2];

  float z2 = zs * zs;
  float j00 = fx / zs, j02 = -fx * xc / z2;
  float j11 = fy / zs, j12 = -fy * yc / z2;
  // u = V*J0, w = V*J1 with J0=(j00,0,j02), J1=(0,j11,j12)
  float u0 = V[0][0] * j00 + V[0][2] * j02;
  float u2 = V[2][0] * j00 + V[2][2] * j02;
  float w0 = V[0][1] * j11 + V[0][2] * j12;
  float w1 = V[1][1] * j11 + V[1][2] * j12;
  float w2 = V[2][1] * j11 + V[2][2] * j12;
  float ca = j00 * u0 + j02 * u2;
  float cb = j00 * w0 + j02 * w2;
  float cc = j11 * w1 + j12 * w2;

  float half_sum = 0.5f * (ca + cc);
  float lam = half_sum + sqrtf(fmaxf(0.25f * (ca - cc) * (ca - cc) + cb * cb, 0.0f));
  float radius = 3.0f * sqrtf(fmaxf(lam, 0.0f));

  bool vis = (zc > 0.1f) && (zc < 100.0f) && (px >= 0.0f) && (px < (float)IMG_W) &&
             (py >= 0.0f) && (py < (float)IMG_H);

  radii_out[i] = vis ? radius : 0.0f;
  ws[WS_KEY + i] = vis ? zc : -1e30f;
  ws[WS_PX + i] = px;  ws[WS_PY + i] = py;
  ws[WS_CA + i] = ca;  ws[WS_CB + i] = cb;  ws[WS_CC + i] = cc;
  ws[WS_Z + i] = zc;   ws[WS_RAD + i] = radius;
  ws[WS_VIS + i] = vis ? 1.0f : 0.0f;
}

// ---------------- Kernel 2: stable descending-z sort + sorted structs + B fragments
__global__ void __launch_bounds__(NG) gs_sort(
    const float* __restrict__ colors, const float* __restrict__ opac,
    float* __restrict__ ws) {
  __shared__ float sKey[NG];
  __shared__ float sCol[NG * 3];
  int i = threadIdx.x;

  float key = ws[WS_KEY + i];
  sKey[i] = key;
  __syncthreads();

  int rank = 0;
  for (int j = 0; j < NG; ++j) {
    float kj = sKey[j];
    rank += (kj > key) || (kj == key && j < i);   // stable, descending
  }

  float a = ws[WS_CA + i], b = ws[WS_CB + i], c = ws[WS_CC + i];
  float ae = a + 1e-6f, ce = c + 1e-6f;
  float inv = 1.0f / (ae * ce - b * b);
  float ia = ce * inv, ib = -b * inv, ic = ae * inv;
  float px = ws[WS_PX + i], py = ws[WS_PY + i];
  float rint = floorf(ws[WS_RAD + i]) + 1.0f;
  float cxp = floorf(px), cyp = floorf(py);

  float4* sg = (float4*)(ws + WS_SG);
  sg[rank * 3 + 0] = make_float4(px, py, ia, ib);
  sg[rank * 3 + 1] = make_float4(ic, opac[i], ws[WS_Z + i], ws[WS_VIS + i]);
  sg[rank * 3 + 2] = make_float4(cxp - rint, cxp + rint, cyp - rint, cyp + rint);

  sCol[rank * 3 + 0] = colors[3 * i + 0];
  sCol[rank * 3 + 1] = colors[3 * i + 1];
  sCol[rank * 3 + 2] = colors[3 * i + 2];
  __syncthreads();

  // Pre-pack WMMA B fragments: B is (32 gaussians) x (16 cols), cols 0..2 = RGB.
  // 16-bit B layout: lane<16 -> N=lane, K in {0..7, 16..23}; lane>=16 -> N=lane-16, K in {8..15, 24..31}.
  int chunk = i >> 5, lane = i & 31;
  int n = lane & 15;
  int kbase = (lane < 16) ? 0 : 8;
  uint32_t* bf = (uint32_t*)ws + WS_BF + ((chunk << 5) + lane) * 8;
#pragma unroll
  for (int d = 0; d < 8; ++d) {
    int k0 = kbase + ((d < 4) ? 2 * d : 2 * d + 8);
    float v0 = (n < 3) ? sCol[((chunk << 5) + k0) * 3 + n] : 0.0f;
    float v1 = (n < 3) ? sCol[((chunk << 5) + k0 + 1) * 3 + n] : 0.0f;
    bf[d] = f2h_bits(v0) | (f2h_bits(v1) << 16);
  }
}

// ---------------- Kernel 3: rasterize; color einsum via v_wmma_f32_16x16x32_f16
__global__ void __launch_bounds__(256) gs_raster(
    const float* __restrict__ ws, float* __restrict__ outRGB,
    float* __restrict__ outDepth, float* __restrict__ outAlpha) {
  __shared__ float4 sh[NG * 3];

  // Async fill of the 24KB sorted-gaussian table: memory -> LDS directly
  // (GLOBAL_LOAD_ASYNC_TO_LDS_B128, GVS mode: mem = saddr + vaddr32).
  {
    const float4* sgbase = (const float4*)(ws + WS_SG);
    uint32_t lds0 = (uint32_t)(uintptr_t)(&sh[0]);   // flat->LDS uses addr[31:0]
    for (int idx = threadIdx.x; idx < NG * 3; idx += 256) {
      uint32_t ldsa = lds0 + (uint32_t)idx * 16u;
      uint32_t voff = (uint32_t)idx * 16u;
      asm volatile("global_load_async_to_lds_b128 %0, %1, %2"
                   :: "v"(ldsa), "v"(voff), "s"(sgbase)
                   : "memory");
    }
    asm volatile("s_wait_asynccnt 0x0" ::: "memory");
  }
  __syncthreads();

  const uint32_t* bfrag = (const uint32_t*)ws + WS_BF;
  int tid = threadIdx.x;
  int lane = tid & 31;
  int pix = blockIdx.x * 256 + tid;          // 300 blocks * 256 == NPIX exactly
  float fgx = (float)(pix % IMG_W);
  float fgy = (float)(pix / IMG_W);

  float T = 1.0f, asum = 0.0f, depth = 0.0f;
  int has = 0;
  v8f c0 = {}, c1 = {};
  uint32_t wp[16];

  for (int chunk = 0; chunk < 16; ++chunk) {
#pragma unroll
    for (int k = 0; k < 32; ++k) {
      int g = (chunk << 5) + k;
      float4 s0 = sh[g * 3 + 0];   // px, py, ia, ib
      float4 s1 = sh[g * 3 + 1];   // ic, op, z, vis
      float4 s2 = sh[g * 3 + 2];   // xmin, xmax, ymin, ymax
      float alpha = 0.0f;
      if (s1.w > 0.5f && fgx >= s2.x && fgx <= s2.y && fgy >= s2.z && fgy <= s2.w) {
        float dx = fgx - s0.x, dy = fgy - s0.y;
        float m = s0.z * dx * dx + 2.0f * s0.w * dx * dy + s1.x * dy * dy;
        alpha = fminf(s1.y * __expf(-0.5f * m), 0.99f);
      }
      float w = alpha * T;
      asum += w;
      if (!has && w > 0.01f) { has = 1; depth = s1.z; }
      T *= (1.0f - alpha);
      uint32_t h = f2h_bits(w);
      if (k & 1) wp[k >> 1] |= (h << 16); else wp[k >> 1] = h;
    }

    // B fragment for this chunk (pre-packed colors)
    V16U bm;
    const uint32_t* bp = bfrag + ((chunk << 5) + lane) * 8;
#pragma unroll
    for (int d = 0; d < 8; ++d) bm.u[d] = bp[d];

    // A fragments (weights), 16x32 f16 layout:
    // lane<16: M=lane,   K in {0..7, 16..23};  lane>=16: M=lane-16, K in {8..15, 24..31}
    int src0 = lane & 15;
    int src1 = src0 + 16;
    V16U a0, a1;
#pragma unroll
    for (int j = 0; j < 8; ++j) {
      int dlo = (j < 4) ? j : (j + 4);           // dwords for K base 0/16
      int lo0 = __shfl((int)wp[dlo], src0, 32);
      int hi0 = __shfl((int)wp[dlo + 4], src0, 32);   // dwords for K base 8/24
      a0.u[j] = (lane < 16) ? (uint32_t)lo0 : (uint32_t)hi0;
      int lo1 = __shfl((int)wp[dlo], src1, 32);
      int hi1 = __shfl((int)wp[dlo + 4], src1, 32);
      a1.u[j] = (lane < 16) ? (uint32_t)lo1 : (uint32_t)hi1;
    }

    c0 = __builtin_amdgcn_wmma_f32_16x16x32_f16(false, a0.v, false, bm.v,
                                                (short)0, c0, false, false);
    c1 = __builtin_amdgcn_wmma_f32_16x16x32_f16(false, a1.v, false, bm.v,
                                                (short)0, c1, false, false);
  }

  outAlpha[pix] = fminf(asum, 1.0f);
  outDepth[pix] = has ? depth : 0.0f;

  // D tile: lanes 0-15 hold N=lane (M=r), lanes 16-31 hold N=lane-16 (M=8+r)
  int laneN = lane & 15;
  int halfSel = lane >> 4;
  int waveBase = blockIdx.x * 256 + (tid & ~31);
  if (laneN < 3) {
#pragma unroll
    for (int r = 0; r < 8; ++r) {
      int p0 = waveBase + halfSel * 8 + r;
      outRGB[p0 * 3 + laneN] = fminf(fmaxf(c0[r], 0.0f), 1.0f);
      int p1 = waveBase + 16 + halfSel * 8 + r;
      outRGB[p1 * 3 + laneN] = fminf(fmaxf(c1[r], 0.0f), 1.0f);
    }
  }
}

extern "C" void kernel_launch(void* const* d_in, const int* in_sizes, int n_in,
                              void* d_out, int out_size, void* d_ws, size_t ws_size,
                              hipStream_t stream) {
  const float* means  = (const float*)d_in[0];
  const float* scales = (const float*)d_in[1];
  const float* rots   = (const float*)d_in[2];
  const float* colors = (const float*)d_in[3];
  const float* opac   = (const float*)d_in[4];
  const float* Km     = (const float*)d_in[5];
  const float* Tm     = (const float*)d_in[6];

  float* out      = (float*)d_out;
  float* outRGB   = out;                       // 230400
  float* outDepth = out + NPIX * 3;            // 76800
  float* outAlpha = out + NPIX * 4;            // 76800
  float* outRadii = out + NPIX * 5;            // 512
  float* ws = (float*)d_ws;

  gs_preproc<<<1, NG, 0, stream>>>(means, scales, rots, Km, Tm, ws, outRadii);
  gs_sort<<<1, NG, 0, stream>>>(colors, opac, ws);
  gs_raster<<<NPIX / 256, 256, 0, stream>>>(ws, outRGB, outDepth, outAlpha);
}
